// Model_82454782148931
// MI455X (gfx1250) — matrix operand order repaired and checked by reading, not compiled, not run on hardware
//
#include <hip/hip_runtime.h>

typedef __bf16 v16bf __attribute__((ext_vector_type(16)));
typedef float  v8f   __attribute__((ext_vector_type(8)));

#define M_DIM  2048
#define K_DIM  4096
#define N_DIM  4096
#define N_REP  30
#define OUT_N  (N_DIM * N_REP)      // 122880

#define BM 128
#define BN 128
#define BK 64
#define KT_STEPS (K_DIM / BK)       // 64

#define A_STRIDE      72            // ushorts per A-tile row (64 + 8 pad) -> 144B, 2-way max
#define B_LANE_STRIDE 40            // ushorts per lane-row in B operand subtile
#define B_TILE        1288          // ushorts per 32x16 B operand subtile (32*40 + 8 pad)

__device__ __forceinline__ unsigned short f2bf(float f) {
  unsigned u = __float_as_uint(f);
  u += 0x7FFFu + ((u >> 16) & 1u);          // round-to-nearest-even
  return (unsigned short)(u >> 16);
}
__device__ __forceinline__ float bf2f(unsigned short h) {
  return __uint_as_float(((unsigned)h) << 16);
}

union OpBF { v16bf v; uint4 q[2]; };

// A operand: lane<16 -> K {0..15}; lane>=16 -> K {16..31}
// p points at row*A_STRIDE + kh*32 + half*16; pick bytes [0..31]
__device__ __forceinline__ v16bf ld_opA(const unsigned short* p) {
  OpBF u;
  const uint4* q = (const uint4*)p;
  u.q[0] = q[0];
  u.q[1] = q[1];
  return u.v;
}
// B operand: lane-row holds 16 contiguous K values (32 bytes)
__device__ __forceinline__ v16bf ld_opB(const unsigned short* p) {
  OpBF u;
  const uint4* q = (const uint4*)p;
  u.q[0] = q[0];
  u.q[1] = q[1];
  return u.v;
}

__global__ void __launch_bounds__(256)
gemm_f32split_tile30(const float* __restrict__ A,
                     const float* __restrict__ B,
                     float* __restrict__ out)
{
  __shared__ unsigned short sAhi[BM * A_STRIDE];    //  18432 B
  __shared__ unsigned short sAlo[BM * A_STRIDE];    //  18432 B
  __shared__ unsigned short sBhi[16 * B_TILE];      //  41216 B
  __shared__ unsigned short sBlo[16 * B_TILE];      //  41216 B  (total ~116.5 KB)

  const int tid  = threadIdx.x;
  const int lane = tid & 31;
  const int wave = tid >> 5;
  const int wm   = wave >> 1;      // 0..3 : 32-row band
  const int wn   = wave & 1;       // 0..1 : 64-col band
  const int half = lane >> 4;
  const int lr   = lane & 15;

  const int bm0 = blockIdx.y * BM;
  const int bn0 = blockIdx.x * BN;

  // --- staging thread mapping ---
  const int arow = tid >> 1;             // 0..127
  const int aseg = (tid & 1) << 5;       // 0 or 32
  const int bk   = tid >> 2;             // k within tile: 0..63
  const int cseg = (tid & 3) << 5;       // n base: 0,32,64,96
  const int jbase = cseg >> 4;           // 0,2,4,6
  const int kc   = bk >> 5;              // K-half subtile: 0/1
  const int k16  = ((bk >> 4) & 1) << 4; // lane-row offset
  const int klo  = bk & 15;              // ushort pos

  const float* gA = A + (size_t)(bm0 + arow) * K_DIM + aseg;
  const float* gB = B + (size_t)bk * N_DIM + bn0 + cseg;

  v8f acc[2][4];
  const v8f vzero = {0.f,0.f,0.f,0.f,0.f,0.f,0.f,0.f};
  #pragma unroll
  for (int i = 0; i < 2; ++i)
    #pragma unroll
    for (int j = 0; j < 4; ++j) acc[i][j] = vzero;

  auto LOAD = [&](float4 (&ra)[8], float4 (&rb)[8], int kt) {
    const float4* pa = (const float4*)(gA + (size_t)kt * BK);
    #pragma unroll
    for (int q = 0; q < 8; ++q) ra[q] = pa[q];
    const float4* pb = (const float4*)(gB + (size_t)kt * BK * N_DIM);
    #pragma unroll
    for (int q = 0; q < 8; ++q) rb[q] = pb[q];
  };

  auto STORE = [&](const float4 (&ra)[8], const float4 (&rb)[8]) {
    // A: row-major bf16 hi/lo, packed pair stores
    unsigned short* ahp = sAhi + arow * A_STRIDE + aseg;
    unsigned short* alp = sAlo + arow * A_STRIDE + aseg;
    #pragma unroll
    for (int q = 0; q < 8; ++q) {
      float4 f = ra[q];
      unsigned short hx = f2bf(f.x), hy = f2bf(f.y), hz = f2bf(f.z), hw = f2bf(f.w);
      unsigned short lx = f2bf(f.x - bf2f(hx)), ly = f2bf(f.y - bf2f(hy));
      unsigned short lz = f2bf(f.z - bf2f(hz)), lw = f2bf(f.w - bf2f(hw));
      ((unsigned*)(ahp + 4*q))[0] = (unsigned)hx | ((unsigned)hy << 16);
      ((unsigned*)(ahp + 4*q))[1] = (unsigned)hz | ((unsigned)hw << 16);
      ((unsigned*)(alp + 4*q))[0] = (unsigned)lx | ((unsigned)ly << 16);
      ((unsigned*)(alp + 4*q))[1] = (unsigned)lz | ((unsigned)lw << 16);
    }
    // B: scatter into WMMA operand order.
    // element (k=bk, n=cseg+e): subtile = (n/16)*2 + k/32,
    //   lane-row = n%16 + 16*((k%32)/16), pos = k%16
    unsigned short* bhp = sBhi + kc * B_TILE + klo;
    unsigned short* blp = sBlo + kc * B_TILE + klo;
    #pragma unroll
    for (int q = 0; q < 8; ++q) {
      float4 f = rb[q];
      #pragma unroll
      for (int c = 0; c < 4; ++c) {
        const float v = (c == 0) ? f.x : (c == 1) ? f.y : (c == 2) ? f.z : f.w;
        const int e  = 4 * q + c;                  // 0..31
        const int jt = jbase + (e >> 4);           // n-tile group
        const int idx = (jt * 2) * B_TILE + ((e & 15) + k16) * B_LANE_STRIDE;
        const unsigned short h = f2bf(v);
        bhp[idx] = h;
        blp[idx] = f2bf(v - bf2f(h));
      }
    }
  };

  auto COMPUTE = [&]() {
    #pragma unroll
    for (int kh = 0; kh < 2; ++kh) {
      v16bf ah[2], al[2];
      #pragma unroll
      for (int i = 0; i < 2; ++i) {
        const int base = (wm * 32 + i * 16 + lr) * A_STRIDE + kh * 32 + half * 16;
        ah[i] = ld_opA(sAhi + base);
        al[i] = ld_opA(sAlo + base);
      }
      #pragma unroll
      for (int j = 0; j < 4; ++j) {
        const int base = ((wn * 4 + j) * 2 + kh) * B_TILE + lane * B_LANE_STRIDE;
        v16bf bh = ld_opB(sBhi + base);
        v16bf bl = ld_opB(sBlo + base);
        #pragma unroll
        for (int i = 0; i < 2; ++i) {
          acc[i][j] = __builtin_amdgcn_wmma_f32_16x16x32_bf16(
              false, ah[i], false, bh, (short)0, acc[i][j], false, false);
          acc[i][j] = __builtin_amdgcn_wmma_f32_16x16x32_bf16(
              false, ah[i], false, bl, (short)0, acc[i][j], false, false);
          acc[i][j] = __builtin_amdgcn_wmma_f32_16x16x32_bf16(
              false, al[i], false, bh, (short)0, acc[i][j], false, false);
        }
      }
    }
  };

  // --- pipelined main loop: single reg set; reload immediately after the LDS
  //     stores consume it, so global loads overlap the 48-WMMA compute phase ---
  float4 ra[8], rb[8];
  LOAD(ra, rb, 0);
  for (int kt = 0; kt < KT_STEPS; ++kt) {
    __syncthreads();                 // previous tile fully consumed by all waves
    STORE(ra, rb);
    if (kt + 1 < KT_STEPS) LOAD(ra, rb, kt + 1);
    __syncthreads();
    COMPUTE();
  }

  // --- epilogue: write the 2048x4096 tile 30x along N, non-temporal ---
  #pragma unroll
  for (int i = 0; i < 2; ++i) {
    #pragma unroll
    for (int j = 0; j < 4; ++j) {
      const int row0 = bm0 + wm * 32 + i * 16 + half * 8;
      const int col  = bn0 + wn * 64 + j * 16 + lr;
      float* pb = out + (size_t)row0 * OUT_N + col;
      #pragma unroll
      for (int v = 0; v < 8; ++v) {
        const float val = acc[i][j][v];
        float* p = pb + (size_t)v * OUT_N;
        #pragma unroll
        for (int r = 0; r < N_REP; ++r)
          __builtin_nontemporal_store(val, p + r * N_DIM);
      }
    }
  }
}

extern "C" void kernel_launch(void* const* d_in, const int* in_sizes, int n_in,
                              void* d_out, int out_size, void* d_ws, size_t ws_size,
                              hipStream_t stream) {
  const float* A = (const float*)d_in[0];   // x1: 2048 x 4096
  const float* B = (const float*)d_in[1];   // x2: 4096 x 4096
  float* out = (float*)d_out;               // 2048 x 122880
  (void)in_sizes; (void)n_in; (void)out_size; (void)d_ws; (void)ws_size;

  dim3 grid(N_DIM / BN, M_DIM / BM);        // 32 x 16 = 512 workgroups
  dim3 block(256);                          // 8 wave32 waves
  gemm_f32split_tile30<<<grid, block, 0, stream>>>(A, B, out);
}
